// PredNoiseAnd3DLUTWR_62302795595974
// MI455X (gfx1250) — matrix operand order repaired and checked by reading, not compiled, not run on hardware
//
#include <hip/hip_runtime.h>
#include <hip/hip_bf16.h>

typedef _Float16 h16;
typedef __attribute__((ext_vector_type(16))) _Float16 v16h;
typedef __attribute__((ext_vector_type(8)))  float    v8f;

#define NJ      539055      // 3*33^3*5
#define KLUT    5
#define DLUT    33
#define D2      1089        // 33*33
#define D3      35937       // 33^3
#define CSTRIDE 35937
#define KSTRIDE 107811      // 3*33^3

// ---------------------------------------------------------------- utilities
__global__ void k_init_scalars(float* s) {
    if (threadIdx.x < 3) s[threadIdx.x] = 0.f;
}

__global__ void k_f32_to_f16(const float* __restrict__ in, h16* __restrict__ out, int n) {
    int i = blockIdx.x * 256 + threadIdx.x;
    if (i < n) out[i] = (h16)in[i];
}

// Pack lb_w1 [256][320][3][3] into B-fragment order:
// wpack[(((kcg*16 + ntile)*32 + lane)*16 + i],  kcg = tap*10 + chunk (K order: k = tap*320 + c)
__global__ void k_pack_w1(const float* __restrict__ w1, h16* __restrict__ wpack) {
    int f = blockIdx.x * 256 + threadIdx.x;
    if (f >= 90 * 16 * 32 * 16) return;
    const int i     = f & 15;
    const int lane  = (f >> 4) & 31;
    const int ntile = (f >> 9) & 15;
    const int kcg   = f >> 13;                  // 0..89
    const int tap   = kcg / 10;
    const int ch    = kcg % 10;
    const int hi    = lane >> 4;
    const int klocal = (i & 7) + ((i >> 3) << 4) + (hi << 3); // B layout: table order k = 8h+s (s<8), 16+8h+(s-8) (s>=8)
    const int c     = (ch << 5) + klocal;
    const int n     = (ntile << 4) + (lane & 15);
    wpack[f] = (h16)w1[(n * 320 + c) * 9 + tap];
}

// ---------------------------------------------------------------- noise_pred conv 3x3 320->4
__global__ void k_conv_out(const float* __restrict__ feat, const float* __restrict__ w,
                           const float* __restrict__ bias, float* __restrict__ out) {
    int idx = blockIdx.x * 256 + threadIdx.x;   // 4*4*32*32
    if (idx >= 16384) return;
    const int x = idx & 31, y = (idx >> 5) & 31, co = (idx >> 10) & 3, b = idx >> 12;
    float acc = bias[co];
    for (int c = 0; c < 320; ++c) {
        const float* fp = &feat[((b * 320 + c) << 10)];
        const float* wp = &w[(co * 320 + c) * 9];
        #pragma unroll
        for (int t = 0; t < 9; ++t) {
            const int yy = y + t / 3 - 1, xx = x + t % 3 - 1;
            if ((unsigned)yy < 32u && (unsigned)xx < 32u)
                acc += fp[(yy << 5) + xx] * wp[t];
        }
    }
    out[idx] = acc;
}

// ---------------------------------------------------------------- lb conv (implicit GEMM, WMMA)
// Block = 256 thr = 8 waves handling (b, ptile, 8 ntiles). Per 3x3 tap we stage ALL 10
// channel-chunks of the A fragment into LDS behind one barrier pair (9 pairs total, not 90),
// then each wave runs 10 ds-fed WMMAs against its pre-packed B fragments.
// h layout: [b][p][n] for coalesced pooling.
__global__ void k_lbconv_wmma(const h16* __restrict__ featH, const h16* __restrict__ wpack,
                              const float* __restrict__ b1, float* __restrict__ hbuf) {
    __shared__ __align__(32) h16 afrag[5120];    // 10 chunks x 32 lanes x 16 halves = 10 KB
    const int tid   = threadIdx.x;
    const int wave  = tid >> 5;
    const int lane  = tid & 31;
    const int nhalf = blockIdx.x & 1;
    const int ptile = (blockIdx.x >> 1) & 63;
    const int b     = blockIdx.x >> 7;
    const int ntile = nhalf * 8 + wave;
    const int n0    = ntile << 4;
    const int p0    = ptile << 4;

    v8f acc = {};
    for (int tap = 0; tap < 9; ++tap) {
        const int dy = tap / 3 - 1, dx = tap % 3 - 1;
        __syncthreads();                          // protect previous tap's LDS reads
        for (int f = tid; f < 5120; f += 256) {   // 20 uniform iterations, 20 indep. loads
            const int chunk = f >> 9;
            const int e  = f & 511;
            const int fl = e >> 4;                // lane slot
            const int i  = e & 15;
            const int m  = fl & 15;
            const int hi2 = fl >> 4;
            const int c  = (chunk << 5) + (i & 7) + ((i >> 3) << 4) + (hi2 << 3);
            const int p  = p0 + m;
            const int yy = (p >> 5) + dy;
            const int xx = (p & 31) + dx;
            h16 v = (h16)0.f;
            if ((unsigned)yy < 32u && (unsigned)xx < 32u)
                v = featH[((b * 320 + c) << 10) + (yy << 5) + xx];
            afrag[f] = v;
        }
        __syncthreads();
        #pragma unroll
        for (int ch = 0; ch < 10; ++ch) {
            const v16h a  = *(const v16h*)&afrag[(ch << 9) + (lane << 4)];
            const int kcg = tap * 10 + ch;
            const v16h bf = *(const v16h*)&wpack[((((kcg << 4) + ntile) << 5) + lane) << 4];
            acc = __builtin_amdgcn_wmma_f32_16x16x32_f16(false, a, false, bf,
                                                         (short)0, acc, false, false);
        }
    }
    // epilogue: +bias, ReLU, store h[b][p][n]
    const int n  = n0 + (lane & 15);
    const int hi = lane >> 4;
    const float bias = b1[n];
    #pragma unroll
    for (int r = 0; r < 8; ++r) {
        const int p = p0 + r + (hi << 3);
        float v = acc[r] + bias;
        hbuf[(((b << 10) + p) << 8) + n] = v > 0.f ? v : 0.f;
    }
}

__global__ void k_pool(const float* __restrict__ hbuf, float* __restrict__ pooled) {
    const int b = blockIdx.x, n = threadIdx.x;
    float s = 0.f;
    for (int p = 0; p < 1024; ++p) s += hbuf[(((b << 10) + p) << 8) + n];
    pooled[(b << 8) + n] = s * (1.f / 1024.f);
}

// ---------------------------------------------------------------- big GEMM (memory-bound, WMMA)
// lut[b][j] = pooled[b] . w2[j] + b2[j].  M=16 (4 real batches), N=16 j's per wave-tile, K=256.
// This streams all 552 MB of lb_w2 exactly once (the roofline of the whole net):
// b128 loads + f16 convert in-register, plus global_prefetch of the wave's next tile.
__global__ void k_gemm_wmma(const float* __restrict__ pooled, const float* __restrict__ w2,
                            const float* __restrict__ b2, float* __restrict__ lut) {
    __shared__ __align__(32) h16 ash[8 * 512];   // A fragments for all 8 K-chunks
    const int tid = threadIdx.x;
    for (int f = tid; f < 4096; f += 256) {
        const int i     = f & 15;
        const int lane  = (f >> 4) & 31;
        const int chunk = f >> 9;
        const int m     = lane & 15;
        const int hi    = lane >> 4;
        const int k     = (chunk << 5) + i + (hi << 4);
        ash[f] = (m < 4) ? (h16)pooled[(m << 8) + k] : (h16)0.f;
    }
    __syncthreads();

    const int lane   = tid & 31;
    const int hi     = lane >> 4;
    const int waveId = blockIdx.x * 8 + (tid >> 5);
    const int nWaves = gridDim.x * 8;
    const int nTiles = (NJ + 15) >> 4;            // 33691

    for (int t = waveId; t < nTiles; t += nWaves) {
        const int j0 = t << 4;
        const int j  = j0 + (lane & 15);
        const int jl = j < NJ ? j : NJ - 1;       // clamp: loads stay in-bounds, store masked

        // Prefetch this wave's NEXT tile (grid-stride), hiding HBM latency behind the WMMAs.
        const int tn = t + nWaves;
        if (tn < nTiles) {
            int jn = tn * 16 + (lane & 15);
            if (jn >= NJ) jn = NJ - 1;
            const float* nb = &w2[(size_t)jn * 256 + (hi << 7)];
            __builtin_prefetch(nb, 0, 1);         // -> global_prefetch_b8
            __builtin_prefetch(nb + 64, 0, 1);
        }

        v8f acc = {};
        #pragma unroll
        for (int chunk = 0; chunk < 8; ++chunk) {
            const v16h a = *(const v16h*)&ash[(chunk << 9) + (lane << 4)];
            const float4* s4 = (const float4*)(w2 + (size_t)jl * 256 + (chunk << 5) + (hi << 4));
            const float4 q0 = s4[0], q1 = s4[1], q2 = s4[2], q3 = s4[3];
            v16h bf;
            bf[0]  = (h16)q0.x; bf[1]  = (h16)q0.y; bf[2]  = (h16)q0.z; bf[3]  = (h16)q0.w;
            bf[4]  = (h16)q1.x; bf[5]  = (h16)q1.y; bf[6]  = (h16)q1.z; bf[7]  = (h16)q1.w;
            bf[8]  = (h16)q2.x; bf[9]  = (h16)q2.y; bf[10] = (h16)q2.z; bf[11] = (h16)q2.w;
            bf[12] = (h16)q3.x; bf[13] = (h16)q3.y; bf[14] = (h16)q3.z; bf[15] = (h16)q3.w;
            acc = __builtin_amdgcn_wmma_f32_16x16x32_f16(false, a, false, bf,
                                                         (short)0, acc, false, false);
        }
        if (hi == 0 && j < NJ) {                  // rows m=0..3 live in lanes 0-15, VGPR 0..3
            const float bias = b2[j];
            #pragma unroll
            for (int r = 0; r < 4; ++r) lut[r * NJ + j] = acc[r] + bias;
        }
    }
}

// ---------------------------------------------------------------- routing network (tiny)
__global__ void k_conv1x1_relu(const float* __restrict__ feat, const float* __restrict__ w,
                               const float* __restrict__ bias, float* __restrict__ out) {
    int idx = blockIdx.x * 256 + threadIdx.x;    // 4*5*32*32
    if (idx >= 20480) return;
    const int pix = idx & 1023;
    const int k = (idx >> 10) % 5, b = (idx >> 10) / 5;
    float acc = bias[k];
    for (int c = 0; c < 320; ++c)
        acc += feat[((b * 320 + c) << 10) + pix] * w[k * 320 + c];
    out[idx] = acc > 0.f ? acc : 0.f;
}

__global__ void k_conv3x3_relu5(const float* __restrict__ in, const float* __restrict__ w,
                                const float* __restrict__ bias, float* __restrict__ out, int S) {
    int idx = blockIdx.x * 256 + threadIdx.x;
    const int per = S * S;
    if (idx >= 20 * per) return;
    const int x = idx % S, y = (idx / S) % S;
    const int k = (idx / per) % 5, b = idx / (5 * per);
    float acc = bias[k];
    for (int c = 0; c < 5; ++c) {
        const float* ip = &in[(b * 5 + c) * per];
        const float* wp = &w[(k * 5 + c) * 9];
        #pragma unroll
        for (int t = 0; t < 9; ++t) {
            const int yy = y + t / 3 - 1, xx = x + t % 3 - 1;
            if ((unsigned)yy < (unsigned)S && (unsigned)xx < (unsigned)S)
                acc += ip[yy * S + xx] * wp[t];
        }
    }
    out[idx] = acc > 0.f ? acc : 0.f;
}

// ConvTranspose2d(k=4, s=2, p=1): w layout [Cin][Cout][4][4]
__global__ void k_convT(const float* __restrict__ in, const float* __restrict__ w,
                        const float* __restrict__ bias, float* __restrict__ out, int Sin) {
    const int So = Sin * 2, per = So * So;
    int idx = blockIdx.x * 256 + threadIdx.x;
    if (idx >= 20 * per) return;
    const int ox = idx % So, oy = (idx / So) % So;
    const int k = (idx / per) % 5, b = idx / (5 * per);
    float acc = bias[k];
    for (int c = 0; c < 5; ++c) {
        const float* ip = &in[(b * 5 + c) * Sin * Sin];
        const float* wp = &w[((c * 5 + k) << 4)];
        #pragma unroll
        for (int ky = 0; ky < 4; ++ky) {
            const int ty = oy + 1 - ky;
            if (ty < 0 || (ty & 1)) continue;
            const int iy = ty >> 1;
            if (iy >= Sin) continue;
            #pragma unroll
            for (int kx = 0; kx < 4; ++kx) {
                const int tx = ox + 1 - kx;
                if (tx < 0 || (tx & 1)) continue;
                const int ix = tx >> 1;
                if (ix >= Sin) continue;
                acc += ip[iy * Sin + ix] * wp[(ky << 2) + kx];
            }
        }
    }
    out[idx] = acc;   // no ReLU after transpose conv
}

// ---------------------------------------------------------------- fused softmax + trilerp + mix
__global__ void k_final(const float* __restrict__ rfin, const float* __restrict__ img,
                        const float* __restrict__ lut, float* __restrict__ oimg,
                        float* __restrict__ wn_acc) {
    int idx = blockIdx.x * 256 + threadIdx.x;    // 4*65536 pixels
    float wn_local = 0.f;
    if (idx < 262144) {
        const int pix = idx & 65535;
        const int b   = idx >> 16;
        float rv[5], mx = -1e30f;
        #pragma unroll
        for (int k = 0; k < 5; ++k) { rv[k] = rfin[(b * 5 + k) * 65536 + pix]; mx = fmaxf(mx, rv[k]); }
        float se = 0.f;
        #pragma unroll
        for (int k = 0; k < 5; ++k) { rv[k] = __expf(rv[k] - mx); se += rv[k]; }
        const float inv = 1.f / se;
        #pragma unroll
        for (int k = 0; k < 5; ++k) { rv[k] *= inv; wn_local += rv[k] * rv[k]; }

        // grid_sample semantics: R->x (last axis), G->y, B->z; align_corners, border clamp
        float fc[3];
        #pragma unroll
        for (int c = 0; c < 3; ++c) {
            float v = img[(b * 3 + c) * 65536 + pix] * 0.5f + 0.5f;
            fc[c] = fminf(fmaxf(v * (float)(DLUT - 1), 0.f), (float)(DLUT - 1));
        }
        int x0 = max(min((int)floorf(fc[0]), DLUT - 2), 0);
        int y0 = max(min((int)floorf(fc[1]), DLUT - 2), 0);
        int z0 = max(min((int)floorf(fc[2]), DLUT - 2), 0);
        const float xd = fc[0] - (float)x0, yd = fc[1] - (float)y0, zd = fc[2] - (float)z0;
        const int base000 = (z0 * DLUT + y0) * DLUT + x0;

        float o[3] = {0.f, 0.f, 0.f};
        for (int k = 0; k < 5; ++k) {
            const float pwk = rv[k];
            const float* bk = &lut[b * NJ + k * KSTRIDE];
            #pragma unroll
            for (int c = 0; c < 3; ++c) {
                const float* L = bk + c * CSTRIDE + base000;
                const float c00 = L[0]       * (1.f - xd) + L[1]        * xd;
                const float c01 = L[DLUT]    * (1.f - xd) + L[DLUT + 1] * xd;
                const float c10 = L[D2]      * (1.f - xd) + L[D2 + 1]   * xd;
                const float c11 = L[D2+DLUT] * (1.f - xd) + L[D2+DLUT+1]* xd;
                const float c0 = c00 * (1.f - yd) + c01 * yd;
                const float c1 = c10 * (1.f - yd) + c11 * yd;
                o[c] += pwk * (c0 * (1.f - zd) + c1 * zd);
            }
        }
        #pragma unroll
        for (int c = 0; c < 3; ++c) oimg[(b * 3 + c) * 65536 + pix] = o[c];
    }
    __shared__ float red[256];
    red[threadIdx.x] = wn_local;
    __syncthreads();
    for (int s = 128; s > 0; s >>= 1) {
        if (threadIdx.x < s) red[threadIdx.x] += red[threadIdx.x + s];
        __syncthreads();
    }
    if (threadIdx.x == 0) atomicAdd(wn_acc, red[0]);
}

// ---------------------------------------------------------------- TV + monotonicity reduction
__global__ void k_reg(const float* __restrict__ lut, float* __restrict__ tv_acc,
                      float* __restrict__ mn_acc) {
    int idx = blockIdx.x * 256 + threadIdx.x;    // 4*5*3*35937
    float tvl = 0.f, mnl = 0.f;
    if (idx < 60 * D3) {
        const int v    = idx % D3;
        const int rest = idx / D3;               // b*15 + k*3 + c
        const int x = v % DLUT, y = (v / DLUT) % DLUT, z = v / D2;
        const float* L = &lut[rest * D3];
        const float v0 = L[v];
        if (x < DLUT - 1) {
            const float d = v0 - L[v + 1];
            const float w = (x == 0 || x == DLUT - 2) ? 2.f : 1.f;
            tvl += d * d * w; mnl += fmaxf(d, 0.f);
        }
        if (y < DLUT - 1) {
            const float d = v0 - L[v + DLUT];
            const float w = (y == 0 || y == DLUT - 2) ? 2.f : 1.f;
            tvl += d * d * w; mnl += fmaxf(d, 0.f);
        }
        if (z < DLUT - 1) {
            const float d = v0 - L[v + D2];
            const float w = (z == 0 || z == DLUT - 2) ? 2.f : 1.f;
            tvl += d * d * w; mnl += fmaxf(d, 0.f);
        }
    }
    __shared__ float rt[256], rm[256];
    rt[threadIdx.x] = tvl; rm[threadIdx.x] = mnl;
    __syncthreads();
    for (int s = 128; s > 0; s >>= 1) {
        if (threadIdx.x < s) { rt[threadIdx.x] += rt[threadIdx.x + s]; rm[threadIdx.x] += rm[threadIdx.x + s]; }
        __syncthreads();
    }
    if (threadIdx.x == 0) { atomicAdd(tv_acc, rt[0]); atomicAdd(mn_acc, rm[0]); }
}

__global__ void k_finalize(const float* __restrict__ s, float* __restrict__ out) {
    // s[0]=sum pw^2, s[1]=tv weighted sq sum, s[2]=mono relu sum
    const float wn = s[0] / (float)(4 * 5 * 65536);
    const float tv = s[1] / (float)(3 * DLUT * DLUT * (DLUT - 1));   // per-(b,k) mean divisor
    const float mn = s[2] / (float)(3 * DLUT * DLUT * (DLUT - 1));
    out[0] = 0.0005f * (wn + tv * 0.25f) + 10.f * mn * 0.25f;        // /B with B=4
}

// ================================================================ launcher
extern "C" void kernel_launch(void* const* d_in, const int* in_sizes, int n_in,
                              void* d_out, int out_size, void* d_ws, size_t ws_size,
                              hipStream_t stream) {
    (void)in_sizes; (void)n_in; (void)out_size; (void)ws_size;
    const float* feat  = (const float*)d_in[0];
    const float* img   = (const float*)d_in[1];
    const float* cow   = (const float*)d_in[2];
    const float* cob   = (const float*)d_in[3];
    const float* lb_w1 = (const float*)d_in[4];
    const float* lb_b1 = (const float*)d_in[5];
    const float* lb_w2 = (const float*)d_in[6];
    const float* lb_b2 = (const float*)d_in[7];
    const float* r_w0  = (const float*)d_in[8];
    const float* r_b0  = (const float*)d_in[9];
    const float* r_w1  = (const float*)d_in[10];
    const float* r_b1  = (const float*)d_in[11];
    const float* r_wt0 = (const float*)d_in[12];
    const float* r_bt0 = (const float*)d_in[13];
    const float* r_w2  = (const float*)d_in[14];
    const float* r_b2  = (const float*)d_in[15];
    const float* r_wt1 = (const float*)d_in[16];
    const float* r_bt1 = (const float*)d_in[17];
    const float* r_w3  = (const float*)d_in[18];
    const float* r_b3  = (const float*)d_in[19];
    const float* r_wt2 = (const float*)d_in[20];
    const float* r_bt2 = (const float*)d_in[21];

    float* out = (float*)d_out;                 // [noise_pred 16384 | lut_image 786432 | reg 1]

    char* ws = (char*)d_ws;
    size_t off = 0;
    auto carve = [&](size_t bytes) -> char* {
        char* p = ws + off;
        off = (off + bytes + 255) & ~(size_t)255;
        return p;
    };
    float* scal   = (float*)carve(3 * sizeof(float));
    h16*   featH  = (h16*)  carve(1310720 * sizeof(h16));
    h16*   wpack  = (h16*)  carve(737280 * sizeof(h16));
    float* hbuf   = (float*)carve(1048576 * sizeof(float));
    float* pooled = (float*)carve(1024 * sizeof(float));
    float* lut    = (float*)carve((size_t)4 * NJ * sizeof(float));
    float* r0     = (float*)carve(20480 * sizeof(float));
    float* r1     = (float*)carve(20480 * sizeof(float));
    float* rt0    = (float*)carve(81920 * sizeof(float));
    float* r2b    = (float*)carve(81920 * sizeof(float));
    float* rt1    = (float*)carve(327680 * sizeof(float));
    float* r3b    = (float*)carve(327680 * sizeof(float));
    float* rfin   = (float*)carve(1310720 * sizeof(float));

    k_init_scalars<<<1, 32, 0, stream>>>(scal);
    k_f32_to_f16<<<5120, 256, 0, stream>>>(feat, featH, 1310720);
    k_pack_w1<<<2880, 256, 0, stream>>>(lb_w1, wpack);

    k_conv_out<<<64, 256, 0, stream>>>(feat, cow, cob, out);

    k_lbconv_wmma<<<512, 256, 0, stream>>>(featH, wpack, lb_b1, hbuf);
    k_pool<<<4, 256, 0, stream>>>(hbuf, pooled);
    k_gemm_wmma<<<512, 256, 0, stream>>>(pooled, lb_w2, lb_b2, lut);

    k_conv1x1_relu<<<80, 256, 0, stream>>>(feat, r_w0, r_b0, r0);
    k_conv3x3_relu5<<<80, 256, 0, stream>>>(r0, r_w1, r_b1, r1, 32);
    k_convT<<<320, 256, 0, stream>>>(r1, r_wt0, r_bt0, rt0, 32);
    k_conv3x3_relu5<<<320, 256, 0, stream>>>(rt0, r_w2, r_b2, r2b, 64);
    k_convT<<<1280, 256, 0, stream>>>(r2b, r_wt1, r_bt1, rt1, 64);
    k_conv3x3_relu5<<<1280, 256, 0, stream>>>(rt1, r_w3, r_b3, r3b, 128);
    k_convT<<<5120, 256, 0, stream>>>(r3b, r_wt2, r_bt2, rfin, 128);

    k_final<<<1024, 256, 0, stream>>>(rfin, img, lut, out + 16384, scal);
    k_reg<<<(60 * D3 + 255) / 256, 256, 0, stream>>>(lut, scal + 1, scal + 2);
    k_finalize<<<1, 1, 0, stream>>>(scal, out + 16384 + 786432);
}
